// CombinedHiddenGCVAEEncoder_16286515987220
// MI455X (gfx1250) — compile-verified, matrix operand order checked by reading, not run on hardware
//
#include <hip/hip_runtime.h>

// ---------------------------------------------------------------------------
// CDNA5 (gfx1250) GCN-VAE encoder. bf16 WMMA GEMMs + f32 atomic aggregation.
// ---------------------------------------------------------------------------

typedef __attribute__((ext_vector_type(16))) __bf16        v16bf;
typedef __attribute__((ext_vector_type(8)))  float         v8f;
typedef __attribute__((ext_vector_type(4)))  unsigned int  u32x4;
typedef __attribute__((ext_vector_type(4)))  float         f32x4;

#define D_INF 512
#define D_HF  512
#define D_ZF  128

// round-to-nearest-even f32 -> bf16 (bit pattern)
__device__ __forceinline__ unsigned short f2bf(float f) {
  unsigned int u = __float_as_uint(f);
  unsigned int r = u + 0x7FFFu + ((u >> 16) & 1u);
  return (unsigned short)(r >> 16);
}

// ---------------------------------------------------------------------------
// degree / normalization
// ---------------------------------------------------------------------------
__global__ void k_fill1(float* __restrict__ p, int n) {
  int i = blockIdx.x * blockDim.x + threadIdx.x;
  if (i < n) p[i] = 1.0f;          // self-loop contributes 1 to every degree
}

__global__ void k_deg_edges(float* __restrict__ deg, const int* __restrict__ dst, int e) {
  int i = blockIdx.x * blockDim.x + threadIdx.x;
  if (i < e) atomicAdd(&deg[dst[i]], 1.0f);
}

__global__ void k_rsqrt_inplace(float* __restrict__ p, int n) {
  int i = blockIdx.x * blockDim.x + threadIdx.x;
  if (i < n) p[i] = rsqrtf(p[i]);  // deg >= 1 always (self loops), no zero guard needed
}

// ---------------------------------------------------------------------------
// conversions
// ---------------------------------------------------------------------------
__global__ void k_f32_to_bf16(const float* __restrict__ in, unsigned short* __restrict__ out,
                              long n) {
  long i = (long)blockIdx.x * blockDim.x + threadIdx.x;
  if (i < n) out[i] = f2bf(in[i]);
}

// Wt[n*K + k] = bf16(W[k*NO + n])  (transpose so WMMA B-fragments load contiguously)
__global__ void k_transpose_bf16(const float* __restrict__ W, unsigned short* __restrict__ Wt,
                                 int K, int NO) {
  int nn = blockIdx.x * 16 + threadIdx.x;  // output-feature index (coalesced read)
  int kk = blockIdx.y * 16 + threadIdx.y;  // input-feature index
  if (nn < NO && kk < K) Wt[(size_t)nn * K + kk] = f2bf(W[(size_t)kk * NO + nn]);
}

// ---------------------------------------------------------------------------
// bf16 WMMA GEMM:  C[M,NO] = A[M,K] (row-major bf16) * Bt[NO,K]^T (bf16)
// One wave computes one 16x16 output tile; K-loop in steps of 32.
// Fragment layout per CDNA5 ISA 7.12.2 (16-bit A 16x32): lane half h = lane/16,
// row = lane%16; the lane's 16 bf16 live at k-offsets [8h, 8h+8) and [16+8h, 16+8h+8),
// i.e. two contiguous 16-byte chunks -> two global_load_b128 per fragment per step.
// ---------------------------------------------------------------------------
union Frag16 { u32x4 u[2]; v16bf v; };

__global__ void k_gemm_bf16(const unsigned short* __restrict__ A,
                            const unsigned short* __restrict__ Bt,
                            float* __restrict__ C,
                            int M, int K, int NO) {
  const int lane   = threadIdx.x & 31;
  const int tilesN = NO >> 4;
  const int tiles  = (M >> 4) * tilesN;
  const int t      = blockIdx.x * blockDim.y + threadIdx.y;   // wave-uniform
  if (t >= tiles) return;                                     // whole wave exits (EXEC stays all-1s)

  const int tm = t / tilesN;
  const int tn = t % tilesN;
  const int hf  = lane >> 4;     // 0 or 1
  const int idx = lane & 15;     // row (A) / col (B)

  const unsigned short* arow = A  + (size_t)(tm * 16 + idx) * K;
  const unsigned short* brow = Bt + (size_t)(tn * 16 + idx) * K;

  v8f acc = {};
  for (int k0 = 0; k0 < K; k0 += 32) {
    Frag16 fa, fb;
    fa.u[0] = *(const u32x4*)(arow + k0 +      8 * hf);
    fa.u[1] = *(const u32x4*)(arow + k0 + 16 + 8 * hf);
    fb.u[0] = *(const u32x4*)(brow + k0 +      8 * hf);
    fb.u[1] = *(const u32x4*)(brow + k0 + 16 + 8 * hf);
    acc = __builtin_amdgcn_wmma_f32_16x16x32_bf16(
        /*neg_a=*/false, fa.v, /*neg_b=*/false, fb.v,
        /*c_mod=*/(short)0, acc, /*reuse_a=*/false, /*reuse_b=*/false);
  }

  // D layout: lane holds col=idx, rows r+8*hf in acc[r]
  float* cp = C + (size_t)(tm * 16 + 8 * hf) * NO + tn * 16 + idx;
#pragma unroll
  for (int r = 0; r < 8; ++r) cp[(size_t)r * NO] = acc[r];
}

// ---------------------------------------------------------------------------
// graph aggregation: out = D^{-1/2}(A+I)D^{-1/2} h
// Self-loop term doubles as the scatter buffer initialization.
// ---------------------------------------------------------------------------
__global__ void k_selfloop_init(const float* __restrict__ h, const float* __restrict__ dinv,
                                float* __restrict__ out, int n, int D) {
  long i = (long)blockIdx.x * blockDim.x + threadIdx.x;
  long total = (long)n * D;
  if (i < total) {
    int node = (int)(i / D);
    float dv = dinv[node];
    out[i] = h[i] * dv * dv;
  }
}

__global__ void k_edge_agg(const float* __restrict__ h, const int* __restrict__ src,
                           const int* __restrict__ dst, const float* __restrict__ dinv,
                           float* __restrict__ out, int e, int D4, int D) {
  for (int ed = blockIdx.x; ed < e; ed += gridDim.x) {
    const int s = src[ed];
    const int d = dst[ed];
    const float nrm = dinv[s] * dinv[d];
    const f32x4* hs = (const f32x4*)(h + (size_t)s * D);
    float* od = out + (size_t)d * D;
    for (int c = threadIdx.x; c < D4; c += blockDim.x) {
      f32x4 v = hs[c];
      atomicAdd(od + 4 * c + 0, v.x * nrm);
      atomicAdd(od + 4 * c + 1, v.y * nrm);
      atomicAdd(od + 4 * c + 2, v.z * nrm);
      atomicAdd(od + 4 * c + 3, v.w * nrm);
    }
  }
}

// ---------------------------------------------------------------------------
// epilogues
// ---------------------------------------------------------------------------
__global__ void k_bias_tanh_bf16(const float* __restrict__ in, const float* __restrict__ b,
                                 unsigned short* __restrict__ out, long n, int D) {
  long i = (long)blockIdx.x * blockDim.x + threadIdx.x;
  if (i < n) out[i] = f2bf(tanhf(in[i] + b[i % D]));
}

__global__ void k_bias_add(const float* __restrict__ in, const float* __restrict__ b,
                           float* __restrict__ out, long n, int D) {
  long i = (long)blockIdx.x * blockDim.x + threadIdx.x;
  if (i < n) out[i] = in[i] + b[i % D];
}

__global__ void k_reparam(const float* __restrict__ noise, const float* __restrict__ mean,
                          const float* __restrict__ logvar, float* __restrict__ z, long n) {
  long i = (long)blockIdx.x * blockDim.x + threadIdx.x;
  if (i < n) z[i] = noise[i] * expf(0.5f * logvar[i]) + mean[i];
}

// ---------------------------------------------------------------------------
// launcher
// ---------------------------------------------------------------------------
extern "C" void kernel_launch(void* const* d_in, const int* in_sizes, int n_in,
                              void* d_out, int out_size, void* d_ws, size_t ws_size,
                              hipStream_t stream) {
  const float* x     = (const float*)d_in[0];
  const int*   ei    = (const int*)d_in[1];     // [2,E] flat (JAX default int32)
  const float* noise = (const float*)d_in[2];
  const float* W1    = (const float*)d_in[3];
  const float* b1    = (const float*)d_in[4];
  const float* W2    = (const float*)d_in[5];
  const float* b2    = (const float*)d_in[6];
  const float* Wm    = (const float*)d_in[7];
  const float* bm    = (const float*)d_in[8];
  const float* Wv    = (const float*)d_in[9];
  const float* bv    = (const float*)d_in[10];

  const int n = in_sizes[0] / D_INF;   // 50000 (multiple of 16)
  const int e = in_sizes[1] / 2;       // 1.6M
  const int* src = ei;
  const int* dst = ei + e;

  // ---- workspace carving (256B aligned) ----
  char* ws = (char*)d_ws;
  size_t off = 0;
  auto carve = [&](size_t bytes) -> char* {
    char* p = ws + off;
    off += (bytes + 255) & ~(size_t)255;
    return p;
  };
  float*          dinv = (float*)carve((size_t)n * sizeof(float));
  float*          bufA = (float*)carve((size_t)n * D_HF * sizeof(float));  // GEMM out / gather src
  float*          bufB = (float*)carve((size_t)n * D_HF * sizeof(float));  // scatter target
  unsigned short* Xbf  = (unsigned short*)carve((size_t)n * D_HF * 2);     // bf16 activations
  unsigned short* Wbf  = (unsigned short*)carve((size_t)D_HF * D_HF * 2);  // bf16 transposed W

  float* zf    = (float*)d_out;
  float* meanf = zf + (size_t)n * D_ZF;
  float* lvf   = meanf + (size_t)n * D_ZF;

  const long nNH = (long)n * D_HF;
  const long nNZ = (long)n * D_ZF;
  auto grid1 = [](long cnt) { return (unsigned)((cnt + 255) / 256); };

  // ---- symmetric normalization ----
  k_fill1<<<grid1(n), 256, 0, stream>>>(dinv, n);
  k_deg_edges<<<grid1(e), 256, 0, stream>>>(dinv, dst, e);
  k_rsqrt_inplace<<<grid1(n), 256, 0, stream>>>(dinv, n);

  // ---- x -> bf16 ----
  k_f32_to_bf16<<<grid1(nNH), 256, 0, stream>>>(x, Xbf, nNH);

  // ---- layer 1: h1 = tanh(Agg(X@W1) + b1) ----
  k_transpose_bf16<<<dim3(D_HF / 16, D_INF / 16), dim3(16, 16), 0, stream>>>(W1, Wbf, D_INF, D_HF);
  {
    int tiles = (n / 16) * (D_HF / 16);
    k_gemm_bf16<<<(tiles + 3) / 4, dim3(32, 4), 0, stream>>>(Xbf, Wbf, bufA, n, D_INF, D_HF);
  }
  k_selfloop_init<<<grid1(nNH), 256, 0, stream>>>(bufA, dinv, bufB, n, D_HF);
  k_edge_agg<<<16384, 128, 0, stream>>>(bufA, src, dst, dinv, bufB, e, D_HF / 4, D_HF);
  k_bias_tanh_bf16<<<grid1(nNH), 256, 0, stream>>>(bufB, b1, Xbf, nNH, D_HF);

  // ---- layer 2: h2 = tanh(Agg(h1@W2) + b2) ----
  k_transpose_bf16<<<dim3(D_HF / 16, D_HF / 16), dim3(16, 16), 0, stream>>>(W2, Wbf, D_HF, D_HF);
  {
    int tiles = (n / 16) * (D_HF / 16);
    k_gemm_bf16<<<(tiles + 3) / 4, dim3(32, 4), 0, stream>>>(Xbf, Wbf, bufA, n, D_HF, D_HF);
  }
  k_selfloop_init<<<grid1(nNH), 256, 0, stream>>>(bufA, dinv, bufB, n, D_HF);
  k_edge_agg<<<16384, 128, 0, stream>>>(bufA, src, dst, dinv, bufB, e, D_HF / 4, D_HF);
  k_bias_tanh_bf16<<<grid1(nNH), 256, 0, stream>>>(bufB, b2, Xbf, nNH, D_HF);

  // ---- mean head: mean = Agg(h2@Wm) + bm ----
  k_transpose_bf16<<<dim3(D_ZF / 16, D_HF / 16), dim3(16, 16), 0, stream>>>(Wm, Wbf, D_HF, D_ZF);
  {
    int tiles = (n / 16) * (D_ZF / 16);
    k_gemm_bf16<<<(tiles + 3) / 4, dim3(32, 4), 0, stream>>>(Xbf, Wbf, bufA, n, D_HF, D_ZF);
  }
  k_selfloop_init<<<grid1(nNZ), 256, 0, stream>>>(bufA, dinv, bufB, n, D_ZF);
  k_edge_agg<<<16384, 32, 0, stream>>>(bufA, src, dst, dinv, bufB, e, D_ZF / 4, D_ZF);
  k_bias_add<<<grid1(nNZ), 256, 0, stream>>>(bufB, bm, meanf, nNZ, D_ZF);

  // ---- logvar head: logvar = Agg(h2@Wv) + bv ----
  k_transpose_bf16<<<dim3(D_ZF / 16, D_HF / 16), dim3(16, 16), 0, stream>>>(Wv, Wbf, D_HF, D_ZF);
  {
    int tiles = (n / 16) * (D_ZF / 16);
    k_gemm_bf16<<<(tiles + 3) / 4, dim3(32, 4), 0, stream>>>(Xbf, Wbf, bufA, n, D_HF, D_ZF);
  }
  k_selfloop_init<<<grid1(nNZ), 256, 0, stream>>>(bufA, dinv, bufB, n, D_ZF);
  k_edge_agg<<<16384, 32, 0, stream>>>(bufA, src, dst, dinv, bufB, e, D_ZF / 4, D_ZF);
  k_bias_add<<<grid1(nNZ), 256, 0, stream>>>(bufB, bv, lvf, nNZ, D_ZF);

  // ---- reparameterization ----
  k_reparam<<<grid1(nNZ), 256, 0, stream>>>(noise, meanf, lvf, zf, nNZ);
}
